// GPT_71494025610072
// MI455X (gfx1250) — compile-verified, matrix operand order checked by reading, not run on hardware
//
#include <hip/hip_runtime.h>

// ---------------- model constants (from reference) ----------------
#define VOCAB  50257
#define DMODEL 768
#define NHEAD  12
#define NLAYER 6
#define HSZ    64
#define FFDIM  3072
#define BSZ    4
#define TSEQ   1024
#define MTOK   (BSZ * TSEQ)   // 4096 token rows
#define LNEPS  1e-5f

typedef long long ll;

// ---------------- WMMA types ----------------
typedef __attribute__((ext_vector_type(16))) __bf16 v16bf;
typedef __attribute__((ext_vector_type(8)))  float  v8f;

struct alignas(16) U128 { unsigned x, y, z, w; };
struct U256 { U128 lo, hi; };          // 32 bytes == v16bf

__device__ __forceinline__ unsigned short f2bfu(float f) {
  unsigned u = __float_as_uint(f);
  unsigned r = u + 0x7FFFu + ((u >> 16) & 1u);   // round-to-nearest-even
  return (unsigned short)(r >> 16);
}

__device__ __forceinline__ float gelu_f(float x) {
  return 0.5f * x * (1.0f + erff(x * 0.70710678118654752f));
}

// CDNA5 async copy: 16B per lane, global -> LDS, tracked by ASYNCcnt.
__device__ __forceinline__ void async_b128(unsigned lds_off, unsigned long long gaddr) {
  asm volatile("global_load_async_to_lds_b128 %0, %1, off"
               :: "v"(lds_off), "v"(gaddr) : "memory");
}
__device__ __forceinline__ void wait_async_le3() {
  asm volatile("s_wait_asynccnt 0x3" ::: "memory");
}
__device__ __forceinline__ void wait_async_0() {
  asm volatile("s_wait_asynccnt 0x0" ::: "memory");
}

// Load one 16x32 bf16 fragment from an LDS tile stored row-major [*][32].
// CDNA5 16-bit A layout: lanes 0-15 hold K[0..8)+K[16..24), lanes 16-31 hold
// K[8..16)+K[24..32) for row M = lane%16 (B fragment identical with tile
// stored N-major, which is how all B operands are laid out in this file).
__device__ __forceinline__ v16bf load_frag(const unsigned short* tile, int row, int koff) {
  const unsigned short* p = tile + row * 32 + koff;
  U256 t;
  t.lo = *(const U128*)(p);
  t.hi = *(const U128*)(p + 16);
  return __builtin_bit_cast(v16bf, t);
}

// ---------------- generic bf16 WMMA GEMM (async double-buffered) ----------------
// C[z][m][n] = act( alpha * sum_k A[z][m][k] * B[z][n][k] + bias[z][n] )
// A: bf16 [M][K] row-major.  B: bf16 [N][K] row-major (pre-transposed weights).
// Outputs: optional f32 Cf (+ optional residual R, same layout) and optional
// bf16 Cb (bmode 0: [m][n] w/ ldcb; bmode 1: V-transpose store [b][n][t]).
// z decomposed as (z1,z2) = (z/zmod, z%zmod); per-tensor offset z1*s1+z2*s2.
#define BM 128
#define BN 64
#define BK 32

__global__ __launch_bounds__(256) void gemm_bf16_wmma(
    const unsigned short* __restrict__ A, ll a_s1, ll a_s2, int lda,
    const unsigned short* __restrict__ B, ll b_s1, ll b_s2, int ldb,
    const float* __restrict__ bias, ll bi_s1, ll bi_s2,
    const float* __restrict__ R,
    float* __restrict__ Cf, ll c_s1, ll c_s2, int ldc,
    unsigned short* __restrict__ Cb, ll cb_s1, ll cb_s2, int ldcb, int bmode,
    int M, int N, int K, int zmod, float alpha, int act)
{
  __shared__ __align__(16) unsigned short lsA[2][BM * BK];   // [m][k]
  __shared__ __align__(16) unsigned short lsB[2][BN * BK];   // [n][k]

  const int z  = blockIdx.z;
  const int z1 = z / zmod, z2 = z % zmod;
  A += z1 * a_s1 + z2 * a_s2;
  B += z1 * b_s1 + z2 * b_s2;
  if (bias) bias += z1 * bi_s1 + z2 * bi_s2;
  if (R)    R    += z1 * c_s1  + z2 * c_s2;
  if (Cf)   Cf   += z1 * c_s1  + z2 * c_s2;
  if (Cb)   Cb   += z1 * cb_s1 + z2 * cb_s2;

  const int m0 = blockIdx.y * BM;
  const int n0 = blockIdx.x * BN;
  const int tid  = threadIdx.x;
  const int lane = tid & 31;
  const int wave = tid >> 5;
  const int wm = (wave >> 1) * 32;        // 4 waves along M
  const int wn = (wave & 1) * 32;         // 2 waves along N
  const bool oob = (n0 + BN > N);

  // Per tile, per wave: 2 async A chunks + 1 async B chunk = 3 instructions.
  auto issue = [&](int kt, int buf) {
    const int k0 = kt * BK;
    for (int i = tid; i < (BM * BK) / 8; i += 256) {       // A: 16B x 512
      int row = i >> 2, seg = i & 3;
      unsigned loff = (unsigned)(uintptr_t)&lsA[buf][row * BK + seg * 8];
      unsigned long long ga =
          (unsigned long long)(uintptr_t)(A + (ll)(m0 + row) * lda + k0 + seg * 8);
      async_b128(loff, ga);
    }
    for (int i = tid; i < (BN * BK) / 8; i += 256) {       // B: 16B x 256
      int row = i >> 2, seg = i & 3;
      int gn = n0 + row; if (gn >= N) gn = N - 1;           // clamp, zero later
      unsigned loff = (unsigned)(uintptr_t)&lsB[buf][row * BK + seg * 8];
      unsigned long long ga =
          (unsigned long long)(uintptr_t)(B + (ll)gn * ldb + k0 + seg * 8);
      async_b128(loff, ga);
    }
  };

  v8f zero = {};
  v8f acc[2][2];
  acc[0][0] = zero; acc[0][1] = zero; acc[1][0] = zero; acc[1][1] = zero;

  const int nt = K / BK;                 // K is always a multiple of 32 here
  issue(0, 0);
  for (int it = 0; it < nt; ++it) {
    const int buf = it & 1;
    // deep prefetch of B two tiles ahead (global_prefetch_b8)
    if (it + 2 < nt && tid < BN) {
      int gn = n0 + tid; if (gn >= N) gn = N - 1;
      __builtin_prefetch(B + (ll)gn * ldb + (it + 2) * BK, 0, 0);
    }
    if (it + 1 < nt) { issue(it + 1, buf ^ 1); wait_async_le3(); }
    else             { wait_async_0(); }
    if (oob) {                            // zero ragged-N rows of retired tile
      U128 zz = {0u, 0u, 0u, 0u};
      for (int i = tid; i < (BN * BK) / 8; i += 256) {
        int row = i >> 2, seg = i & 3;
        if (n0 + row >= N) *(U128*)&lsB[buf][row * BK + seg * 8] = zz;
      }
    }
    __syncthreads();                      // tile[buf] visible to all waves

    const int r    = lane & 15;
    const int koff = (lane >> 4) << 3;
    v16bf a0 = load_frag(&lsA[buf][0], wm + r,      koff);
    v16bf a1 = load_frag(&lsA[buf][0], wm + 16 + r, koff);
    v16bf b0 = load_frag(&lsB[buf][0], wn + r,      koff);
    v16bf b1 = load_frag(&lsB[buf][0], wn + 16 + r, koff);

    acc[0][0] = __builtin_amdgcn_wmma_f32_16x16x32_bf16(false, a0, false, b0, (short)0, acc[0][0], false, false);
    acc[0][1] = __builtin_amdgcn_wmma_f32_16x16x32_bf16(false, a0, false, b1, (short)0, acc[0][1], false, false);
    acc[1][1] = __builtin_amdgcn_wmma_f32_16x16x32_bf16(false, a1, false, b1, (short)0, acc[1][1], false, false);
    acc[1][0] = __builtin_amdgcn_wmma_f32_16x16x32_bf16(false, a1, false, b0, (short)0, acc[1][0], false, false);
    __syncthreads();                      // all reads done before buf reuse
  }

  // C/D layout: VGPR e -> M = e + (lane>=16 ? 8 : 0), N = lane%16
  const int cn  = lane & 15;
  const int cmb = (lane >> 4) << 3;
#pragma unroll
  for (int i = 0; i < 2; ++i) {
#pragma unroll
    for (int j = 0; j < 2; ++j) {
      int gn = n0 + wn + j * 16 + cn;
      if (gn < N) {
        float bv = bias ? bias[gn] : 0.0f;
#pragma unroll
        for (int e = 0; e < 8; ++e) {
          int gm = m0 + wm + i * 16 + cmb + e;
          float val = alpha * acc[i][j][e] + bv;
          if (act == 1) val = gelu_f(val);
          if (Cf) {
            ll ci = (ll)gm * ldc + gn;
            float o = val;
            if (R) o += R[ci];
            Cf[ci] = o;
          }
          if (Cb) {
            ll bi2;
            if (bmode == 0) bi2 = (ll)gm * ldcb + gn;
            else  // V-transpose: [b][hs][t], b = gm/T, t = gm%T
              bi2 = (ll)(gm >> 10) * ((ll)DMODEL * TSEQ) + (ll)gn * TSEQ + (gm & (TSEQ - 1));
            Cb[bi2] = f2bfu(val);
          }
        }
      }
    }
  }
}

// ---------------- weight transpose + f32->bf16 convert ----------------
// src: f32 [K][N] (slab blockIdx.z, stride K*N) -> dst: bf16 [N][K]
__global__ __launch_bounds__(256) void tconv_k(const float* __restrict__ src,
                                               unsigned short* __restrict__ dst,
                                               int K, int N)
{
  __shared__ float t[32][33];
  ll slab = (ll)blockIdx.z * K * N;
  src += slab; dst += slab;
  int kb0 = blockIdx.y * 32, nb0 = blockIdx.x * 32;
  int tn = threadIdx.x & 31, tk = threadIdx.x >> 5;   // tk: 0..7
#pragma unroll
  for (int i = 0; i < 4; ++i) {
    int k = kb0 + tk + i * 8, n = nb0 + tn;
    t[tk + i * 8][tn] = (k < K && n < N) ? src[(ll)k * N + n] : 0.0f;
  }
  __syncthreads();
#pragma unroll
  for (int i = 0; i < 4; ++i) {
    int n = nb0 + tk + i * 8, k = kb0 + tn;
    if (n < N && k < K) dst[(ll)n * K + k] = f2bfu(t[tn][tk + i * 8]);
  }
}

// ---------------- embedding (f32 residual stream) ----------------
__global__ __launch_bounds__(256) void embed_k(const int* __restrict__ idx,
                                               const float* __restrict__ tok,
                                               const float* __restrict__ pos,
                                               float* __restrict__ x)
{
  int bt = blockIdx.x;
  int t  = bt & (TSEQ - 1);
  ll to = (ll)idx[bt] * DMODEL;
  for (int d = threadIdx.x; d < DMODEL; d += 256)
    x[(ll)bt * DMODEL + d] = tok[to + d] + pos[(ll)t * DMODEL + d];
}

// ---------------- layernorm: f32 in -> bf16 out ----------------
__global__ __launch_bounds__(256) void layernorm_k(const float* __restrict__ x,
                                                   const float* __restrict__ s,
                                                   const float* __restrict__ b,
                                                   unsigned short* __restrict__ y)
{
  __shared__ float red[256];
  int row = blockIdx.x;
  const float* xr = x + (ll)row * DMODEL;
  float sum = 0.0f;
  for (int d = threadIdx.x; d < DMODEL; d += 256) sum += xr[d];
  red[threadIdx.x] = sum; __syncthreads();
  for (int o = 128; o > 0; o >>= 1) {
    if (threadIdx.x < o) red[threadIdx.x] += red[threadIdx.x + o];
    __syncthreads();
  }
  float mean = red[0] / (float)DMODEL;
  __syncthreads();
  float vs = 0.0f;
  for (int d = threadIdx.x; d < DMODEL; d += 256) { float t = xr[d] - mean; vs += t * t; }
  red[threadIdx.x] = vs; __syncthreads();
  for (int o = 128; o > 0; o >>= 1) {
    if (threadIdx.x < o) red[threadIdx.x] += red[threadIdx.x + o];
    __syncthreads();
  }
  float inv = rsqrtf(red[0] / (float)DMODEL + LNEPS);
  unsigned short* yr = y + (ll)row * DMODEL;
  for (int d = threadIdx.x; d < DMODEL; d += 256)
    yr[d] = f2bfu((xr[d] - mean) * inv * s[d] + b[d]);
}

// ---------------- causal softmax: f32 scores -> bf16 probs ----------------
__global__ __launch_bounds__(256) void softmax_causal_k(const float* __restrict__ S,
                                                        unsigned short* __restrict__ P)
{
  __shared__ float red[256];
  ll z = blockIdx.y;
  int t = blockIdx.x;
  const float* row = S + z * (ll)TSEQ * TSEQ + (ll)t * TSEQ;
  unsigned short* pr = P + z * (ll)TSEQ * TSEQ + (ll)t * TSEQ;
  float mx = -3.0e38f;
  for (int s = threadIdx.x; s <= t; s += 256) mx = fmaxf(mx, row[s]);
  red[threadIdx.x] = mx; __syncthreads();
  for (int o = 128; o > 0; o >>= 1) {
    if (threadIdx.x < o) red[threadIdx.x] = fmaxf(red[threadIdx.x], red[threadIdx.x + o]);
    __syncthreads();
  }
  mx = red[0]; __syncthreads();
  float sum = 0.0f;
  for (int s = threadIdx.x; s <= t; s += 256) sum += expf(row[s] - mx);
  red[threadIdx.x] = sum; __syncthreads();
  for (int o = 128; o > 0; o >>= 1) {
    if (threadIdx.x < o) red[threadIdx.x] += red[threadIdx.x + o];
    __syncthreads();
  }
  float inv = 1.0f / red[0];
  for (int s = threadIdx.x; s <= t; s += 256) pr[s] = f2bfu(expf(row[s] - mx) * inv);
  for (int s = t + 1 + threadIdx.x; s < TSEQ; s += 256) pr[s] = 0;
}

// ---------------- cross-entropy loss ----------------
__global__ void zero_loss_k(float* p) { if (threadIdx.x == 0) p[0] = 0.0f; }

__global__ __launch_bounds__(256) void loss_k(const float* __restrict__ logits,
                                              const int* __restrict__ targets,
                                              float* __restrict__ loss)
{
  __shared__ float red[256];
  int row = blockIdx.x;
  const float* lr = logits + (ll)row * VOCAB;
  float mx = -3.0e38f;
  for (int i = threadIdx.x; i < VOCAB; i += 256) mx = fmaxf(mx, lr[i]);
  red[threadIdx.x] = mx; __syncthreads();
  for (int o = 128; o > 0; o >>= 1) {
    if (threadIdx.x < o) red[threadIdx.x] = fmaxf(red[threadIdx.x], red[threadIdx.x + o]);
    __syncthreads();
  }
  mx = red[0]; __syncthreads();
  float sum = 0.0f;
  for (int i = threadIdx.x; i < VOCAB; i += 256) sum += expf(lr[i] - mx);
  red[threadIdx.x] = sum; __syncthreads();
  for (int o = 128; o > 0; o >>= 1) {
    if (threadIdx.x < o) red[threadIdx.x] += red[threadIdx.x + o];
    __syncthreads();
  }
  if (threadIdx.x == 0) {
    float lse = mx + logf(red[0]);
    atomicAdd(loss, (lse - lr[targets[row]]) * (1.0f / (float)MTOK));
  }
}

// ---------------- host side ----------------
static inline void gemm(hipStream_t st,
                        const unsigned short* A, ll a1, ll a2, int lda,
                        const unsigned short* B, ll b1, ll b2, int ldb,
                        const float* bias, ll bi1, ll bi2,
                        const float* R,
                        float* Cf, ll c1, ll c2, int ldc,
                        unsigned short* Cb, ll cb1, ll cb2, int ldcb, int bmode,
                        int M, int N, int K, int Z, int zmod, float alpha, int act)
{
  dim3 grid((N + BN - 1) / BN, M / BM, Z);
  gemm_bf16_wmma<<<grid, dim3(256), 0, st>>>(A, a1, a2, lda, B, b1, b2, ldb,
                                             bias, bi1, bi2, R, Cf, c1, c2, ldc,
                                             Cb, cb1, cb2, ldcb, bmode,
                                             M, N, K, zmod, alpha, act);
}

extern "C" void kernel_launch(void* const* d_in, const int* in_sizes, int n_in,
                              void* d_out, int out_size, void* d_ws, size_t ws_size,
                              hipStream_t stream)
{
  (void)in_sizes; (void)n_in; (void)out_size; (void)ws_size;
  const int*   idx   = (const int*)  d_in[0];
  const int*   tgt   = (const int*)  d_in[1];
  const float* tok   = (const float*)d_in[2];
  const float* pos   = (const float*)d_in[3];
  const float* Wq    = (const float*)d_in[4];
  const float* bq    = (const float*)d_in[5];
  const float* Wk    = (const float*)d_in[6];
  const float* bk    = (const float*)d_in[7];
  const float* Wv    = (const float*)d_in[8];
  const float* bv    = (const float*)d_in[9];
  const float* Wproj = (const float*)d_in[10];
  const float* bproj = (const float*)d_in[11];
  const float* W1    = (const float*)d_in[12];
  const float* b1    = (const float*)d_in[13];
  const float* W2    = (const float*)d_in[14];
  const float* b2    = (const float*)d_in[15];
  const float* ln1s  = (const float*)d_in[16];
  const float* ln1b  = (const float*)d_in[17];
  const float* ln2s  = (const float*)d_in[18];
  const float* ln2b  = (const float*)d_in[19];
  const float* lnfs  = (const float*)d_in[20];
  const float* lnfb  = (const float*)d_in[21];
  const float* Whead = (const float*)d_in[22];

  float* logits = (float*)d_out;
  float* loss   = logits + (ll)MTOK * VOCAB;

  const ll MD = (ll)MTOK * DMODEL;             // 3.1M
  const ll TT = (ll)TSEQ * TSEQ;
  const ll ZTT = (ll)BSZ * NHEAD * TT;         // 50.3M
  const ll TD = (ll)TSEQ * DMODEL;

  // f32 region: residual stream + raw scores
  float* x = (float*)d_ws;                     // MD
  float* S = x + MD;                           // ZTT
  // bf16 region: activations then transposed weights
  unsigned short* hb   = (unsigned short*)(S + ZTT);
  unsigned short* qb   = hb   + MD;
  unsigned short* kb   = qb   + MD;
  unsigned short* vT   = kb   + MD;            // [B][H][HS][T]
  unsigned short* attb = vT   + MD;
  unsigned short* ffbb = attb + MD;            // MTOK*FF
  unsigned short* Pb   = ffbb + (ll)MTOK * FFDIM;          // bf16 probs, ZTT
  unsigned short* Wqt  = Pb   + ZTT;           // [L][H][HS][D]
  unsigned short* Wkt  = Wqt  + (ll)NLAYER * DMODEL * DMODEL;
  unsigned short* Wvt  = Wkt  + (ll)NLAYER * DMODEL * DMODEL;
  unsigned short* Wpt  = Wvt  + (ll)NLAYER * DMODEL * DMODEL;   // [L][D][D]
  unsigned short* W1t  = Wpt  + (ll)NLAYER * DMODEL * DMODEL;   // [L][FF][D]
  unsigned short* W2t  = W1t  + (ll)NLAYER * DMODEL * FFDIM;    // [L][D][FF]
  unsigned short* Wht  = W2t  + (ll)NLAYER * DMODEL * FFDIM;    // [V][D]

  // one-time (per launch) weight transpose-convert f32 [K][N] -> bf16 [N][K]
  auto tc = [&](const float* s, unsigned short* d, int K, int N, int slabs) {
    dim3 g((N + 31) / 32, (K + 31) / 32, slabs);
    tconv_k<<<g, dim3(256), 0, stream>>>(s, d, K, N);
  };
  tc(Wq, Wqt, DMODEL, HSZ, NLAYER * NHEAD);
  tc(Wk, Wkt, DMODEL, HSZ, NLAYER * NHEAD);
  tc(Wv, Wvt, DMODEL, HSZ, NLAYER * NHEAD);
  tc(Wproj, Wpt, DMODEL, DMODEL, NLAYER);
  tc(W1, W1t, DMODEL, FFDIM, NLAYER);
  tc(W2, W2t, FFDIM, DMODEL, NLAYER);
  tc(Whead, Wht, DMODEL, VOCAB, 1);

  const float scale = 1.0f / sqrtf((float)DMODEL);   // reference scales by D

  embed_k<<<MTOK, 256, 0, stream>>>(idx, tok, pos, x);

  for (int l = 0; l < NLAYER; ++l) {
    layernorm_k<<<MTOK, 256, 0, stream>>>(x, ln1s + l * DMODEL, ln1b + l * DMODEL, hb);
    const ll wslab = (ll)HSZ * DMODEL;
    // Q/K/V head projections (grid.z = head); V stored transposed [b][hs][t]
    gemm(stream, hb, 0, 0, DMODEL,
         Wqt + (ll)l * NHEAD * wslab, 0, wslab, DMODEL,
         bq + (ll)l * NHEAD * HSZ, 0, HSZ, nullptr,
         nullptr, 0, 0, DMODEL,
         qb, 0, HSZ, DMODEL, 0,
         MTOK, HSZ, DMODEL, NHEAD, NHEAD, 1.0f, 0);
    gemm(stream, hb, 0, 0, DMODEL,
         Wkt + (ll)l * NHEAD * wslab, 0, wslab, DMODEL,
         bk + (ll)l * NHEAD * HSZ, 0, HSZ, nullptr,
         nullptr, 0, 0, DMODEL,
         kb, 0, HSZ, DMODEL, 0,
         MTOK, HSZ, DMODEL, NHEAD, NHEAD, 1.0f, 0);
    gemm(stream, hb, 0, 0, DMODEL,
         Wvt + (ll)l * NHEAD * wslab, 0, wslab, DMODEL,
         bv + (ll)l * NHEAD * HSZ, 0, HSZ, nullptr,
         nullptr, 0, 0, DMODEL,
         vT, 0, (ll)HSZ * TSEQ, TSEQ, 1 /*v-transpose*/,
         MTOK, HSZ, DMODEL, NHEAD, NHEAD, 1.0f, 0);
    // S = scale * Q @ K^T   (B operand = kb, naturally [n=s][k=hs])
    gemm(stream, qb, TD, HSZ, DMODEL,
         kb, TD, HSZ, DMODEL,
         nullptr, 0, 0, nullptr,
         S, (ll)NHEAD * TT, TT, TSEQ,
         nullptr, 0, 0, TSEQ, 0,
         TSEQ, TSEQ, HSZ, BSZ * NHEAD, NHEAD, scale, 0);
    softmax_causal_k<<<dim3(TSEQ, BSZ * NHEAD), 256, 0, stream>>>(S, Pb);
    // att = P @ V            (B operand = vT, [n=hs][k=s])
    gemm(stream, Pb, (ll)NHEAD * TT, TT, TSEQ,
         vT, (ll)DMODEL * TSEQ, (ll)HSZ * TSEQ, TSEQ,
         nullptr, 0, 0, nullptr,
         nullptr, 0, 0, DMODEL,
         attb, TD, HSZ, DMODEL, 0,
         TSEQ, HSZ, TSEQ, BSZ * NHEAD, NHEAD, 1.0f, 0);
    // x = x + att @ Wproj + bproj
    gemm(stream, attb, 0, 0, DMODEL,
         Wpt + (ll)l * DMODEL * DMODEL, 0, 0, DMODEL,
         bproj + (ll)l * DMODEL, 0, 0, x,
         x, 0, 0, DMODEL,
         nullptr, 0, 0, DMODEL, 0,
         MTOK, DMODEL, DMODEL, 1, 1, 1.0f, 0);
    // FF
    layernorm_k<<<MTOK, 256, 0, stream>>>(x, ln2s + l * DMODEL, ln2b + l * DMODEL, hb);
    gemm(stream, hb, 0, 0, DMODEL,
         W1t + (ll)l * DMODEL * FFDIM, 0, 0, DMODEL,
         b1 + (ll)l * FFDIM, 0, 0, nullptr,
         nullptr, 0, 0, FFDIM,
         ffbb, 0, 0, FFDIM, 0,
         MTOK, FFDIM, DMODEL, 1, 1, 1.0f, 1 /*gelu*/);
    gemm(stream, ffbb, 0, 0, FFDIM,
         W2t + (ll)l * DMODEL * FFDIM, 0, 0, FFDIM,
         b2 + (ll)l * DMODEL, 0, 0, x,
         x, 0, 0, DMODEL,
         nullptr, 0, 0, DMODEL, 0,
         MTOK, DMODEL, FFDIM, 1, 1, 1.0f, 0);
  }

  layernorm_k<<<MTOK, 256, 0, stream>>>(x, lnfs, lnfb, hb);
  // logits = h @ Whead (N = 50257, ragged tiles zero-filled)
  gemm(stream, hb, 0, 0, DMODEL,
       Wht, 0, 0, DMODEL,
       nullptr, 0, 0, nullptr,
       logits, 0, 0, VOCAB,
       nullptr, 0, 0, VOCAB, 0,
       MTOK, VOCAB, DMODEL, 1, 1, 1.0f, 0);

  zero_loss_k<<<1, 32, 0, stream>>>(loss);
  loss_k<<<MTOK, 256, 0, stream>>>(logits, tgt, loss);
}
